// InfoNCE_32719060861359
// MI455X (gfx1250) — compile-verified, matrix operand order checked by reading, not run on hardware
//
#include <hip/hip_runtime.h>
#include <hip/hip_bf16.h>
#include <math.h>

// ---------------------------------------------------------------------------
// InfoNCE loss: B=8192, N=8192, D=1024, fp32 in, fp32 scalar out.
// Strategy: bf16 WMMA GEMM (q @ n^T) fused with exp/row-sum epilogue.
// Double-buffered LDS, branch-free steady-state loop (last chunk peeled),
// one barrier per K-chunk, v_rcp epilogue.
// ---------------------------------------------------------------------------

typedef __attribute__((ext_vector_type(16))) __bf16 v16bf;
typedef __attribute__((ext_vector_type(4)))  __bf16 v4bf;
typedef __attribute__((ext_vector_type(8)))  float  v8f;

#define TEMP      0.92f
#define INV_T     (1.0f / TEMP)
#define EPS       1e-6f
#define DCONST    1024
#define TILE      128
#define KC        32
#define LDST      40   // padded LDS row stride (bf16 elems): conflict-free reads

union AFrag { v16bf v; v4bf q[4]; };

// Load one 16x32 bf16 fragment for this lane.
// rowp points at the LDS row for this lane's M (or N) index; kb is 0 or 8.
// Register map (ISA 7.12.2, 16-bit A 16x32): elems 0..7 = K kb..kb+7,
// elems 8..15 = K kb+16..kb+23.  Adjacent v4bf pairs fuse into ds_load_b128.
__device__ inline void load_frag(const __bf16* rowp, int kb, AFrag& f) {
    const v4bf* p0 = (const v4bf*)(rowp + kb);
    const v4bf* p1 = (const v4bf*)(rowp + kb + 16);
    f.q[0] = p0[0];
    f.q[1] = p0[1];
    f.q[2] = p1[0];
    f.q[3] = p1[1];
}

__device__ inline v4bf cvt4(float4 f) {
    v4bf r;
    r.x = (__bf16)f.x; r.y = (__bf16)f.y; r.z = (__bf16)f.z; r.w = (__bf16)f.w;
    return r;
}

// ---------------------------------------------------------------------------
// Kernel 1: per-row norms, positive similarity, s2 zero-init.
// Blocks [0,B): q/p rows.  Blocks [B,B+N): negative rows.
// ---------------------------------------------------------------------------
__global__ __launch_bounds__(256) void norms_kernel(
    const float* __restrict__ Q, const float* __restrict__ P,
    const float* __restrict__ Nk,
    float* __restrict__ qn, float* __restrict__ nn,
    float* __restrict__ s1, float* __restrict__ s2, int B, int D)
{
    __shared__ float red[3][8];
    const int b = blockIdx.x;
    const int t = threadIdx.x;
    const int w = t >> 5, l = t & 31;

    if (b < B) {
        const float4* q4 = (const float4*)(Q + (size_t)b * D);
        const float4* p4 = (const float4*)(P + (size_t)b * D);
        float sq = 0.f, sp = 0.f, dp = 0.f;
        for (int i = t; i < D / 4; i += 256) {
            float4 q = q4[i], p = p4[i];
            sq += q.x*q.x + q.y*q.y + q.z*q.z + q.w*q.w;
            sp += p.x*p.x + p.y*p.y + p.z*p.z + p.w*p.w;
            dp += q.x*p.x + q.y*p.y + q.z*p.z + q.w*p.w;
        }
        for (int off = 16; off > 0; off >>= 1) {
            sq += __shfl_down(sq, off, 32);
            sp += __shfl_down(sp, off, 32);
            dp += __shfl_down(dp, off, 32);
        }
        if (l == 0) { red[0][w] = sq; red[1][w] = sp; red[2][w] = dp; }
        __syncthreads();
        if (t == 0) {
            float SQ = 0.f, SP = 0.f, DP = 0.f;
            #pragma unroll
            for (int i = 0; i < 8; ++i) { SQ += red[0][i]; SP += red[1][i]; DP += red[2][i]; }
            float nq = sqrtf(SQ), np = sqrtf(SP);
            qn[b] = nq;
            s1[b] = __expf(DP / fmaxf(nq * np, EPS) * INV_T);
            s2[b] = 0.0f;   // re-init accumulator every launch (deterministic)
        }
    } else {
        const int n = b - B;
        const float4* n4 = (const float4*)(Nk + (size_t)n * D);
        float sn = 0.f;
        for (int i = t; i < D / 4; i += 256) {
            float4 v = n4[i];
            sn += v.x*v.x + v.y*v.y + v.z*v.z + v.w*v.w;
        }
        for (int off = 16; off > 0; off >>= 1) sn += __shfl_down(sn, off, 32);
        if (l == 0) red[0][w] = sn;
        __syncthreads();
        if (t == 0) {
            float S = 0.f;
            #pragma unroll
            for (int i = 0; i < 8; ++i) S += red[0][i];
            nn[n] = sqrtf(S);
        }
    }
}

// ---------------------------------------------------------------------------
// Kernel 2: S = q @ n^T, fused eps-clamped cosine + exp(/T) + row-sum into s2.
// 128x128 tile per block; 8 wave32s in a 4x2 grid, each wave 32x64 via
// 8x v_wmma_f32_16x16x32_bf16 per K-chunk of 32.  Ping-pong LDS buffers,
// last K-chunk peeled so the steady-state loop is branch-free.
// ---------------------------------------------------------------------------
__global__ __launch_bounds__(256) void gemm_exp_rowsum(
    const float* __restrict__ Q, const float* __restrict__ Nk,
    const float* __restrict__ qn, const float* __restrict__ nn,
    float* __restrict__ s2, int D)
{
    __shared__ __bf16 sA[2][TILE * LDST];
    __shared__ __bf16 sB[2][TILE * LDST];

    const int tid  = threadIdx.x;
    const int lane = tid & 31;
    const int w    = tid >> 5;
    const int wm   = w >> 1;           // 0..3 -> 32-row strip
    const int wn   = w & 1;            // 0..1 -> 64-col strip
    const int rowBase = blockIdx.y * TILE;
    const int colBase = blockIdx.x * TILE;

    // cooperative load mapping: thread -> (tile row, half of 32-wide K chunk)
    const int lr = tid >> 1;
    const int lh = (tid & 1) * 16;

    const int m16 = lane & 15;         // fragment M / N index
    const int kb  = (lane < 16) ? 0 : 8;

    const float* gaBase = Q  + (size_t)(rowBase + lr) * D + lh;
    const float* gbBase = Nk + (size_t)(colBase + lr) * D + lh;
    const int ldsOff = lr * LDST + lh;

    v8f zero = {};
    v8f acc[2][4];
    #pragma unroll
    for (int i = 0; i < 2; ++i)
        #pragma unroll
        for (int j = 0; j < 4; ++j) acc[i][j] = zero;

    // ---- prologue: stage chunk 0 into buffer 0 ----
    #pragma unroll
    for (int i = 0; i < 4; ++i) {
        float4 fa4 = ((const float4*)gaBase)[i];
        float4 fb4 = ((const float4*)gbBase)[i];
        *(v4bf*)(&sA[0][ldsOff + i * 4]) = cvt4(fa4);
        *(v4bf*)(&sB[0][ldsOff + i * 4]) = cvt4(fb4);
    }
    __syncthreads();

    // ---- steady state: branch-free; computes chunks [0, D-KC) ----
    int buf = 0;
    for (int k0 = 0; k0 < D - KC; k0 += KC, buf ^= 1) {
        // prefetch chunk k0+KC from global (L2-resident) into registers
        v4bf na[4], nb[4];
        {
            const float* ga = gaBase + k0 + KC;
            const float* gb = gbBase + k0 + KC;
            #pragma unroll
            for (int i = 0; i < 4; ++i) {
                na[i] = cvt4(((const float4*)ga)[i]);
                nb[i] = cvt4(((const float4*)gb)[i]);
            }
        }

        // fragments from current buffer
        AFrag fa[2], fb[4];
        #pragma unroll
        for (int i = 0; i < 2; ++i)
            load_frag(&sA[buf][(wm * 32 + i * 16 + m16) * LDST], kb, fa[i]);
        #pragma unroll
        for (int j = 0; j < 4; ++j)
            load_frag(&sB[buf][(wn * 64 + j * 16 + m16) * LDST], kb, fb[j]);

        #pragma unroll
        for (int i = 0; i < 2; ++i)
            #pragma unroll
            for (int j = 0; j < 4; ++j)
                acc[i][j] = __builtin_amdgcn_wmma_f32_16x16x32_bf16(
                    false, fa[i].v, false, fb[j].v,
                    (short)0, acc[i][j], false, false);

        // stage next chunk into the other buffer; safe because the barrier
        // at the end of the previous iteration ordered all reads of buf^1.
        #pragma unroll
        for (int i = 0; i < 4; ++i) {
            *(v4bf*)(&sA[buf ^ 1][ldsOff + i * 4]) = na[i];
            *(v4bf*)(&sB[buf ^ 1][ldsOff + i * 4]) = nb[i];
        }
        __syncthreads();
    }

    // ---- peeled final chunk: compute only, no staging ----
    {
        AFrag fa[2], fb[4];
        #pragma unroll
        for (int i = 0; i < 2; ++i)
            load_frag(&sA[buf][(wm * 32 + i * 16 + m16) * LDST], kb, fa[i]);
        #pragma unroll
        for (int j = 0; j < 4; ++j)
            load_frag(&sB[buf][(wn * 64 + j * 16 + m16) * LDST], kb, fb[j]);

        #pragma unroll
        for (int i = 0; i < 2; ++i)
            #pragma unroll
            for (int j = 0; j < 4; ++j)
                acc[i][j] = __builtin_amdgcn_wmma_f32_16x16x32_bf16(
                    false, fa[i].v, false, fb[j].v,
                    (short)0, acc[i][j], false, false);
    }

    // Epilogue: sim = dot / max(qn*nn, eps); e = exp(sim/T); row-sum -> s2.
    // rcp is ample precision: denom ~O(1e3), error ~1 ulp vs bf16 dot noise.
    const int half = lane >> 4;        // C/D layout: reg e holds M = e + 8*half
    float nnv[4];
    #pragma unroll
    for (int j = 0; j < 4; ++j)
        nnv[j] = nn[colBase + wn * 64 + j * 16 + m16];

    #pragma unroll
    for (int i = 0; i < 2; ++i) {
        #pragma unroll
        for (int e = 0; e < 8; ++e) {
            const int row = rowBase + wm * 32 + i * 16 + e + 8 * half;
            const float qv = qn[row];
            float s = 0.f;
            #pragma unroll
            for (int j = 0; j < 4; ++j) {
                float rinv = __builtin_amdgcn_rcpf(fmaxf(qv * nnv[j], EPS));
                s += __expf(acc[i][j][e] * INV_T * rinv);
            }
            // butterfly over the 16-lane half (cols 0..15 of this tile row)
            #pragma unroll
            for (int off = 1; off < 16; off <<= 1)
                s += __shfl_xor(s, off, 32);
            if (m16 == 0)
                atomicAdd(&s2[row], s);
        }
    }
}

// ---------------------------------------------------------------------------
// Kernel 3: loss = mean_b( -log(s1 / (s1 + s2/N)) )
// ---------------------------------------------------------------------------
__global__ __launch_bounds__(256) void loss_kernel(
    const float* __restrict__ s1, const float* __restrict__ s2,
    float* __restrict__ out, int B, int N)
{
    __shared__ float red[8];
    const int t = threadIdx.x;
    const float invN = 1.0f / (float)N;
    float acc = 0.f;
    for (int b = t; b < B; b += 256) {
        float a = s1[b];
        float m = s2[b] * invN;
        acc += -__logf(a / (a + m));
    }
    for (int off = 16; off > 0; off >>= 1) acc += __shfl_down(acc, off, 32);
    if ((t & 31) == 0) red[t >> 5] = acc;
    __syncthreads();
    if (t == 0) {
        float S = 0.f;
        #pragma unroll
        for (int i = 0; i < 8; ++i) S += red[i];
        out[0] = S / (float)B;
    }
}

// ---------------------------------------------------------------------------
extern "C" void kernel_launch(void* const* d_in, const int* in_sizes, int n_in,
                              void* d_out, int out_size, void* d_ws, size_t ws_size,
                              hipStream_t stream) {
    const float* Q  = (const float*)d_in[0];  // [B, 1, D]
    const float* P  = (const float*)d_in[1];  // [B, 1, D]
    const float* Nk = (const float*)d_in[2];  // [N, 1, D]
    const int D = DCONST;
    const int B = in_sizes[0] / D;
    const int N = in_sizes[2] / D;

    float* ws = (float*)d_ws;
    float* qn = ws;            // [B]
    float* nn = qn + B;        // [N]
    float* s1 = nn + N;        // [B]
    float* s2 = s1 + B;        // [B]

    norms_kernel<<<B + N, 256, 0, stream>>>(Q, P, Nk, qn, nn, s1, s2, B, D);

    dim3 grid(N / TILE, B / TILE);
    gemm_exp_rowsum<<<grid, 256, 0, stream>>>(Q, Nk, qn, nn, s2, D);

    loss_kernel<<<1, 256, 0, stream>>>(s1, s2, (float*)d_out, B, N);
}
